// Seq2SeqA_64639257805373
// MI455X (gfx1250) — matrix operand order repaired and checked by reading, not compiled, not run on hardware
//
#include <hip/hip_runtime.h>
#include <stdint.h>
#include <stddef.h>

#define L_ 256
#define B_ 64
#define V_ 32000
#define E_ 512
#define H_ 1024
#define LDS_STRIDE 1032   // 1024 + 8 bf16 pad -> row stride 2064B, conflict-free frag reads

typedef __bf16 bf16_t;
typedef __attribute__((ext_vector_type(16))) __bf16        v16bf;
typedef __attribute__((ext_vector_type(8)))  float         v8f;
typedef __attribute__((ext_vector_type(4)))  unsigned int  u32x4;

union FragBF {
    struct { u32x4 lo; u32x4 hi; } u;
    v16bf v;
};

// A-matrix 16x32 bf16 fragment: lanes 0-15 hold row M=lane with
// K=0..15 contiguously; lanes 16-31 hold K=16..31 (matches load_frag_b).
__device__ __forceinline__ v16bf load_frag_a(const bf16_t* __restrict__ tile, int lda, int lane) {
    const int m    = lane & 15;
    const int half = lane >> 4;
    const bf16_t* p = tile + (size_t)m * lda + half * 16;
    FragBF f;
    f.u.lo = *(const u32x4*)(p);
    f.u.hi = *(const u32x4*)(p + 8);
    return f.v;
}

// B-matrix 32x16 bf16 fragment from row-major U (B(k,n)=U[n][k]): lane holds
// column N=lane&15; lanes 0-15 carry K=0..15 contiguously, lanes 16-31 K=16..31.
__device__ __forceinline__ v16bf load_frag_b(const bf16_t* __restrict__ tile, int ldb, int lane) {
    const int n    = lane & 15;
    const int half = lane >> 4;
    const bf16_t* p = tile + (size_t)n * ldb + half * 16;
    FragBF f;
    f.u.lo = *(const u32x4*)(p);
    f.u.hi = *(const u32x4*)(p + 8);
    return f.v;
}

__device__ __forceinline__ v8f wmma_bf16(v16bf a, v16bf b, v8f c) {
    return __builtin_amdgcn_wmma_f32_16x16x32_bf16(false, a, false, b, (short)0, c, false, false);
}

__device__ __forceinline__ float sigmoidf_(float x) { return 1.0f / (1.0f + __expf(-x)); }

// ---------------------------------------------------------------- utilities
__global__ void zero_f32(float* p, size_t n) {
    size_t i = (size_t)blockIdx.x * blockDim.x + threadIdx.x;
    const size_t stride = (size_t)gridDim.x * blockDim.x;
    for (; i < n; i += stride) p[i] = 0.0f;
}

__global__ void cvt_bf16(const float* __restrict__ s, bf16_t* __restrict__ d, int n) {
    int i = blockIdx.x * blockDim.x + threadIdx.x;
    const int stride = gridDim.x * blockDim.x;
    for (; i < n; i += stride) d[i] = (bf16_t)s[i];
}

// Embedding gather for ltr and flipped (rtl) sequences -> bf16 activations.
__global__ void gather_embed(const int* __restrict__ tokens, const int* __restrict__ lengths,
                             const float* __restrict__ emb,
                             bf16_t* __restrict__ x_ltr, bf16_t* __restrict__ x_rtl) {
    const int i = blockIdx.x;            // 0 .. L*B-1, i = t*B + b
    const int t = i / B_;
    const int b = i - t * B_;
    const int tl  = tokens[t * B_ + b];
    const int idx = lengths[b] - 1 - t;
    const int tr  = (idx >= 0) ? tokens[idx * B_ + b] : 0;   // flip_padded -> token 0 in padding
    for (int e = threadIdx.x; e < E_; e += blockDim.x) {
        x_ltr[(size_t)i * E_ + e] = (bf16_t)emb[(size_t)tl * E_ + e];
        x_rtl[(size_t)i * E_ + e] = (bf16_t)emb[(size_t)tr * E_ + e];
    }
}

// X_g = x * W_g^T + b_g for the 3 gates of one direction.
__global__ void __launch_bounds__(128)
xgemm(const bf16_t* __restrict__ x,
      const bf16_t* __restrict__ W0, const bf16_t* __restrict__ W1, const bf16_t* __restrict__ W2,
      const float* __restrict__ b0, const float* __restrict__ b1, const float* __restrict__ b2,
      float* __restrict__ X0, float* __restrict__ X1, float* __restrict__ X2) {
    const int lane = threadIdx.x & 31;
    const int wave = threadIdx.x >> 5;
    const int row0 = blockIdx.x * 16;
    const int n0   = (blockIdx.y * 4 + wave) * 16;
    v8f c0 = {}, c1 = {}, c2 = {};
    for (int k0 = 0; k0 < E_; k0 += 32) {
        v16bf a  = load_frag_a(x  + (size_t)row0 * E_ + k0, E_, lane);
        v16bf f0 = load_frag_b(W0 + (size_t)n0   * E_ + k0, E_, lane);
        v16bf f1 = load_frag_b(W1 + (size_t)n0   * E_ + k0, E_, lane);
        v16bf f2 = load_frag_b(W2 + (size_t)n0   * E_ + k0, E_, lane);
        c0 = wmma_bf16(a, f0, c0);
        c1 = wmma_bf16(a, f1, c1);
        c2 = wmma_bf16(a, f2, c2);
    }
    const int n  = n0 + (lane & 15);
    const int mb = (lane >> 4) * 8;
    const float s0 = b0[n], s1 = b1[n], s2 = b2[n];
    #pragma unroll
    for (int v = 0; v < 8; ++v) {
        const size_t o = (size_t)(row0 + mb + v) * H_ + n;
        X0[o] = c0[v] + s0;
        X1[o] = c1[v] + s1;
        X2[o] = c2[v] + s2;
    }
}

// Device-scope grid barrier: per-wave release fence, one-arrival atomic,
// s_sleep polling, then acquire fence to invalidate stale L0/L1 lines.
__device__ __forceinline__ void grid_barrier(unsigned* cnt, unsigned target) {
    __threadfence();                         // each wave flushes its stores to device scope
    __syncthreads();
    if (threadIdx.x == 0) {
        atomicAdd(cnt, 1u);
        while (atomicAdd(cnt, 0u) < target)  // L2-coherent poll
            __builtin_amdgcn_s_sleep(1);
    }
    __syncthreads();
    __threadfence();                         // invalidate before reading peers' h writes
}

// Persistent bidirectional GRU recurrence: one launch for all 256 steps.
// grid (H/16, 2 dirs), block 128 (4 waves = 4 row tiles of B=64).
// The block's 3-gate U column slab (3 x 16 x 1024 bf16, ~97KB) lives in LDS
// for the whole kernel; only the bf16 h mirror crosses workgroups per step.
__global__ void __launch_bounds__(128)
gru_persist(bf16_t* __restrict__ hb, float* __restrict__ hf,
            const bf16_t* __restrict__ Uall, const float* __restrict__ Xall,
            const int* __restrict__ lengths, float* __restrict__ out,
            unsigned* __restrict__ bar) {
    extern __shared__ __align__(16) char smem_raw[];
    bf16_t* ldsU = (bf16_t*)smem_raw;        // [3][16][LDS_STRIDE]

    const int dir  = blockIdx.y;
    const int lane = threadIdx.x & 31;
    const int wave = threadIdx.x >> 5;
    const int n0   = blockIdx.x * 16;
    const int row0 = wave * 16;
    const unsigned nblk = gridDim.x * gridDim.y;
    const size_t HH  = (size_t)H_ * H_;
    const size_t BH  = (size_t)B_ * H_;
    const size_t LBH = (size_t)L_ * B_ * H_;

    // one-time: stage U tiles (3 gates x 16 output cols x full K) into LDS
    for (int idx = threadIdx.x; idx < 3 * 16 * (H_ / 8); idx += blockDim.x) {
        const int g  = idx >> 11;            // / (16*128)
        const int r  = idx & 2047;
        const int n  = r >> 7;
        const int kv = r & 127;
        *(u32x4*)&ldsU[((size_t)g * 16 + n) * LDS_STRIDE + kv * 8] =
            *(const u32x4*)(Uall + ((size_t)dir * 3 + g) * HH + (size_t)(n0 + n) * H_ + kv * 8);
    }
    __syncthreads();

    const int nl = lane & 15;
    const int half = lane >> 4;
    const int n  = n0 + nl;
    const int mb = row0 + half * 8;
    const float* Xb[3];
    #pragma unroll
    for (int g = 0; g < 3; ++g) Xb[g] = Xall + ((size_t)dir * 3 + g) * LBH;
    const size_t ldsB = ((size_t)nl) * LDS_STRIDE + half * 16;   // per-lane B base offset

    for (int t = 0; t < L_; ++t) {
        const int cur = t & 1, nxt = cur ^ 1;
        const bf16_t* hin = hb + ((size_t)dir * 2 + cur) * BH;
        const float* Xh = Xb[0] + (size_t)t * BH;
        const float* Xz = Xb[1] + (size_t)t * BH;
        const float* Xr = Xb[2] + (size_t)t * BH;

        // prefetch this step's gate pre-activations while the K loop runs
        __builtin_prefetch(Xh + (size_t)mb * H_ + n, 0, 1);
        __builtin_prefetch(Xz + (size_t)mb * H_ + n, 0, 1);
        __builtin_prefetch(Xr + (size_t)mb * H_ + n, 0, 1);

        v8f gh = {}, gz = {}, gr = {};
        for (int k0 = 0; k0 < H_; k0 += 32) {
            v16bf a = load_frag_a(hin + (size_t)row0 * H_ + k0, H_, lane);
            FragBF f0, f1, f2;
            const bf16_t* p0 = ldsU + 0 * 16 * LDS_STRIDE + ldsB + k0;
            const bf16_t* p1 = ldsU + 1 * 16 * LDS_STRIDE + ldsB + k0;
            const bf16_t* p2 = ldsU + 2 * 16 * LDS_STRIDE + ldsB + k0;
            f0.u.lo = *(const u32x4*)p0; f0.u.hi = *(const u32x4*)(p0 + 8);
            f1.u.lo = *(const u32x4*)p1; f1.u.hi = *(const u32x4*)(p1 + 8);
            f2.u.lo = *(const u32x4*)p2; f2.u.hi = *(const u32x4*)(p2 + 8);
            gh = wmma_bf16(a, f0.v, gh);
            gz = wmma_bf16(a, f1.v, gz);
            gr = wmma_bf16(a, f2.v, gr);
        }

        const float* hfin  = hf + ((size_t)dir * 2 + cur) * BH;
        float*       hfout = hf + ((size_t)dir * 2 + nxt) * BH;
        bf16_t*      hbout = hb + ((size_t)dir * 2 + nxt) * BH;

        #pragma unroll
        for (int v = 0; v < 8; ++v) {
            const int b   = mb + v;
            const int len = lengths[b];
            const size_t o = (size_t)b * H_ + n;
            const float hprev = hfin[o];
            const float r  = sigmoidf_(Xr[o] + gr[v]);
            const float z  = sigmoidf_(Xz[o] + gz[v]);
            const float hp = tanhf(Xh[o] + r * gh[v]);
            const float hn = z * hp + (1.0f - z) * hprev;
            const bool valid = t < len;
            const float hnew = valid ? hn : hprev;
            hfout[o] = hnew;
            hbout[o] = (bf16_t)hnew;
            if (valid) {
                const int l = (dir == 0) ? t : (len - 1 - t);   // rtl writes through the flip
                out[(size_t)l * (B_ * 2 * H_) + (size_t)b * (2 * H_) + (size_t)dir * H_ + n] = hn;
            }
        }
        grid_barrier(bar, nblk * (unsigned)(t + 1));
    }
}

// dec_init = h_last * Ws_w^T + Ws_b.
__global__ void __launch_bounds__(128)
dec_gemm(const bf16_t* __restrict__ hlast, const bf16_t* __restrict__ Ws,
         const float* __restrict__ Wsb, float* __restrict__ out) {
    const int lane = threadIdx.x & 31;
    const int wave = threadIdx.x >> 5;
    const int row0 = wave * 16;
    const int n0   = blockIdx.x * 16;
    v8f c = {};
    for (int k0 = 0; k0 < H_; k0 += 32) {
        v16bf a = load_frag_a(hlast + (size_t)row0 * H_ + k0, H_, lane);
        v16bf b = load_frag_b(Ws    + (size_t)n0   * H_ + k0, H_, lane);
        c = wmma_bf16(a, b, c);
    }
    const int n  = n0 + (lane & 15);
    const int mb = row0 + (lane >> 4) * 8;
    const float bias = Wsb[n];
    #pragma unroll
    for (int v = 0; v < 8; ++v)
        out[(size_t)(mb + v) * H_ + n] = c[v] + bias;
}

extern "C" void kernel_launch(void* const* d_in, const int* in_sizes, int n_in,
                              void* d_out, int out_size, void* d_ws, size_t ws_size,
                              hipStream_t stream) {
    (void)in_sizes; (void)n_in; (void)ws_size;
    const int*   tokens  = (const int*)d_in[0];
    const int*   lengths = (const int*)d_in[1];
    const float* emb     = (const float*)d_in[2];
    const float* W_f32[6] = { (const float*)d_in[3],  (const float*)d_in[4],  (const float*)d_in[5],
                              (const float*)d_in[12], (const float*)d_in[13], (const float*)d_in[14] };
    const float* U_f32[6] = { (const float*)d_in[6],  (const float*)d_in[7],  (const float*)d_in[8],
                              (const float*)d_in[15], (const float*)d_in[16], (const float*)d_in[17] };
    const float* bias[6]  = { (const float*)d_in[9],  (const float*)d_in[10], (const float*)d_in[11],
                              (const float*)d_in[18], (const float*)d_in[19], (const float*)d_in[20] };
    const float* Wsw = (const float*)d_in[21];
    const float* Wsb = (const float*)d_in[22];
    float* out = (float*)d_out;

    // ---- workspace layout (~460 MB) ----
    char* ws = (char*)d_ws;
    size_t off = 0;
    bf16_t* x_ltr = (bf16_t*)(ws + off); off += (size_t)L_ * B_ * E_ * 2;
    bf16_t* x_rtl = (bf16_t*)(ws + off); off += (size_t)L_ * B_ * E_ * 2;
    bf16_t* Wbf   = (bf16_t*)(ws + off); off += (size_t)6 * H_ * E_ * 2;
    bf16_t* Ubf   = (bf16_t*)(ws + off); off += (size_t)6 * H_ * H_ * 2;
    bf16_t* Wsbf  = (bf16_t*)(ws + off); off += (size_t)H_ * H_ * 2;
    float*  Xg    = (float*)(ws + off);  off += (size_t)6 * L_ * B_ * H_ * 4;
    float*  hf    = (float*)(ws + off);  off += (size_t)4 * B_ * H_ * 4;     // [dir][buf]
    bf16_t* hb    = (bf16_t*)(ws + off); off += (size_t)4 * B_ * H_ * 2;     // [dir][buf]
    unsigned* bar = (unsigned*)(ws + off); off += 256;                        // grid barrier

    // zero output (padding slots are never written), hidden state, barrier counter
    zero_f32<<<dim3(2048), 256, 0, stream>>>(out, (size_t)out_size);
    zero_f32<<<dim3(64),   256, 0, stream>>>(hf, (size_t)4 * B_ * H_);
    zero_f32<<<dim3(32),   256, 0, stream>>>((float*)hb, (size_t)4 * B_ * H_ / 2);
    zero_f32<<<dim3(1),    64,  0, stream>>>((float*)bar, 64);

    // weights -> bf16 (one-time)
    for (int g = 0; g < 6; ++g)
        cvt_bf16<<<dim3(512), 256, 0, stream>>>(W_f32[g], Wbf + (size_t)g * H_ * E_, H_ * E_);
    for (int g = 0; g < 6; ++g)
        cvt_bf16<<<dim3(1024), 256, 0, stream>>>(U_f32[g], Ubf + (size_t)g * H_ * H_, H_ * H_);
    cvt_bf16<<<dim3(1024), 256, 0, stream>>>(Wsw, Wsbf, H_ * H_);

    gather_embed<<<dim3(L_ * B_), 256, 0, stream>>>(tokens, lengths, emb, x_ltr, x_rtl);

    // parallel gate pre-projections (103 GFLOP of bf16 WMMA)
    for (int dir = 0; dir < 2; ++dir) {
        xgemm<<<dim3(L_ * B_ / 16, H_ / 64), 128, 0, stream>>>(
            dir == 0 ? x_ltr : x_rtl,
            Wbf + (size_t)(dir * 3 + 0) * H_ * E_,
            Wbf + (size_t)(dir * 3 + 1) * H_ * E_,
            Wbf + (size_t)(dir * 3 + 2) * H_ * E_,
            bias[dir * 3 + 0], bias[dir * 3 + 1], bias[dir * 3 + 2],
            Xg + (size_t)(dir * 3 + 0) * L_ * B_ * H_,
            Xg + (size_t)(dir * 3 + 1) * L_ * B_ * H_,
            Xg + (size_t)(dir * 3 + 2) * L_ * B_ * H_);
    }

    // persistent recurrence: 128 workgroups (fit comfortably at 3 per WGP with
    // 97KB LDS each on a >=43-WGP part), all 256 steps in one launch
    const size_t lds_bytes = (size_t)3 * 16 * LDS_STRIDE * sizeof(bf16_t);
    gru_persist<<<dim3(H_ / 16, 2), 128, lds_bytes, stream>>>(hb, hf, Ubf, Xg, lengths, out, bar);

    // after t=255 the live buffer is index 0; ltr final state == h[len-1] (mask freezes h)
    dec_gemm<<<dim3(H_ / 16), 128, 0, stream>>>(hb /* dir0,buf0 */, Wsbf, Wsb,
                                                out + (size_t)L_ * B_ * 2 * H_);
}
